// NeuronControlActor_26517128086196
// MI455X (gfx1250) — matrix operand order repaired and checked by reading, not compiled, not run on hardware
//
#include <hip/hip_runtime.h>
#include <math.h>

// ---------------- model dimensions (match reference) ----------------
#define STATE_SZ   1024
#define D_MODEL    1024
#define D_STATE    64
#define D_CONV     4
#define HEADDIM    64
#define D_INNER    2048      // EXPAND * D_MODEL
#define NHEADS     32        // D_INNER / HEADDIM
#define CONV_DIM   2176      // D_INNER + 2*D_STATE
#define D_IN_PROJ  4256      // 2*D_INNER + 2*D_STATE + NHEADS
#define ZXP        4352      // D_IN_PROJ padded to multiple of 256
#define BATCH      2
#define SEQLEN     512
#define MROWS      (BATCH*SEQLEN)   // 1024 token rows

// ---------------- WMMA types ----------------
typedef __attribute__((ext_vector_type(16))) __bf16 v16bf;
typedef __attribute__((ext_vector_type(8)))  float  v8f;
typedef __attribute__((ext_vector_type(2)))  int    i2v;

// ---------------- gfx1250 async global->LDS path (guarded) ----------
#if defined(__has_builtin)
#if __has_builtin(__builtin_amdgcn_global_load_async_to_lds_b64) && \
    __has_builtin(__builtin_amdgcn_s_wait_asynccnt)
#define USE_ASYNC_LDS 1
#endif
#endif
#ifndef USE_ASYNC_LDS
#define USE_ASYNC_LDS 0
#endif

// builtin signature (from hipcc diagnostic): pointers to v2i32 in AS1/AS3
__device__ static inline __attribute__((address_space(1))) i2v*
as_global_b64(const void* p) {
  return (__attribute__((address_space(1))) i2v*)p;
}
__device__ static inline __attribute__((address_space(3))) i2v*
as_lds_b64(void* p) {
  return (__attribute__((address_space(3))) i2v*)p;
}

// =====================================================================
// GEMM: C[M,Npad] = act(A[M,K](bf16) @ Wt[Npad,K](bf16, pre-transposed))
// block tile 32(M) x 256(N); 8 waves; wave (wm,wn) -> 16x64 via 4 wmma
// accumulators reusing one A fragment. Double-buffered LDS, async
// global->LDS copies overlapped with WMMA compute.
// act: 0 = none, 1 = relu, 2 = sigmoid.  Writes f32 and/or bf16 output.
// All dims padded: no bounds checks anywhere.
// =====================================================================
#define BM 32
#define BN 256
#define BK 32

__global__ __launch_bounds__(256) void gemm_wmma(
    const __bf16* __restrict__ A, const __bf16* __restrict__ Wt,
    const float* __restrict__ bias, float* __restrict__ Cf,
    __bf16* __restrict__ Cb, int M, int N, int K, int act)
{
  __shared__ __bf16 sA[2][BM][BK + 8];    // [buf][m][k]
  __shared__ __bf16 sB[2][BN][BK + 8];    // [buf][n][k]  (K-contiguous)

  const int tid  = threadIdx.x;
  const int lane = tid & 31;
  const int w    = tid >> 5;      // wave 0..7
  const int wm   = w >> 2;        // 0..1
  const int wn   = w & 3;         // 0..3
  const int m0   = blockIdx.y * BM;
  const int n0   = blockIdx.x * BN;

  v8f acc[4];
#pragma unroll
  for (int s = 0; s < 4; ++s)
#pragma unroll
    for (int r = 0; r < 8; ++r) acc[s][r] = 0.0f;

  // stage tile kt into LDS buffer buf (pure bf16 byte copy)
  auto stage = [&](int kt, int buf) {
    const int k0 = kt * BK;
    {   // A tile: 32 rows x 64B; 8 threads/row, 8B each
      const int row = tid >> 3;
      const int seg = tid & 7;
      const __bf16* g = A + (long)(m0 + row) * K + k0 + seg * 4;
      __bf16* l = &sA[buf][row][seg * 4];
#if USE_ASYNC_LDS
      __builtin_amdgcn_global_load_async_to_lds_b64(
          as_global_b64(g), as_lds_b64(l), 0, 0);
#else
      *(uint2*)l = *(const uint2*)g;
#endif
    }
    {   // B tile: 256 rows x 64B; 1 thread/row
      const __bf16* g = Wt + (long)(n0 + tid) * K + k0;
      __bf16* l = &sB[buf][tid][0];
#if USE_ASYNC_LDS
      __builtin_amdgcn_global_load_async_to_lds_b64(as_global_b64(g), as_lds_b64(l), 0, 0);
      __builtin_amdgcn_global_load_async_to_lds_b64(as_global_b64(g), as_lds_b64(l), 8, 0);
      __builtin_amdgcn_global_load_async_to_lds_b64(as_global_b64(g), as_lds_b64(l), 16, 0);
      __builtin_amdgcn_global_load_async_to_lds_b64(as_global_b64(g), as_lds_b64(l), 24, 0);
      __builtin_amdgcn_global_load_async_to_lds_b64(as_global_b64(g), as_lds_b64(l), 32, 0);
      __builtin_amdgcn_global_load_async_to_lds_b64(as_global_b64(g), as_lds_b64(l), 40, 0);
      __builtin_amdgcn_global_load_async_to_lds_b64(as_global_b64(g), as_lds_b64(l), 48, 0);
      __builtin_amdgcn_global_load_async_to_lds_b64(as_global_b64(g), as_lds_b64(l), 56, 0);
#else
#pragma unroll
      for (int i = 0; i < 4; ++i)
        ((uint4*)l)[i] = ((const uint4*)g)[i];
#endif
    }
  };

  const int nk = K / BK;
  stage(0, 0);

  for (int kt = 0; kt < nk; ++kt) {
    const int cur = kt & 1;
#if USE_ASYNC_LDS
    __builtin_amdgcn_s_wait_asynccnt(0);   // own async copies landed
#endif
    __syncthreads();                       // all waves' tiles visible
    if (kt + 1 < nk) stage(kt + 1, cur ^ 1);   // overlap next copy

    // ---- A fragment (16x32 bf16, contiguous k per lane half, same as B): lane -> row lane%16,
    //      K = klo..klo+15, klo=(lane/16)*16 ----
    v16bf afrag;
    {
      const int arow = wm * 16 + (lane & 15);
      const int klo  = (lane >> 4) * 16;
#pragma unroll
      for (int e = 0; e < 8; ++e) {
        afrag[e]     = sA[cur][arow][klo + e];
        afrag[e + 8] = sA[cur][arow][klo + 8 + e];
      }
    }
    // ---- 4 x (B fragment + WMMA), reusing A ----
    const int kb = (lane >> 4) * 16;
#pragma unroll
    for (int sub = 0; sub < 4; ++sub) {
      v16bf bfrag;
      const int nl = wn * 64 + sub * 16 + (lane & 15);
#pragma unroll
      for (int e = 0; e < 16; ++e) bfrag[e] = sB[cur][nl][kb + e];
      acc[sub] = __builtin_amdgcn_wmma_f32_16x16x32_bf16(
          false, afrag, false, bfrag, (short)0, acc[sub], false, false);
    }
  }

  // ---- epilogue: C/D layout row=(lane/16)*8+r, col=lane%16 ----
  const int rbase = m0 + wm * 16 + (lane >> 4) * 8;
  const int cbase = n0 + wn * 64 + (lane & 15);
#pragma unroll
  for (int sub = 0; sub < 4; ++sub) {
#pragma unroll
    for (int r = 0; r < 8; ++r) {
      const int row = rbase + r;
      const int col = cbase + sub * 16;
      float v = acc[sub][r];
      if (bias) v += bias[col];
      if (act == 1)      v = fmaxf(v, 0.0f);
      else if (act == 2) v = 1.0f / (1.0f + __expf(-v));
      if (Cf) Cf[(long)row * N + col] = v;
      if (Cb) Cb[(long)row * N + col] = (__bf16)v;
    }
  }
}

// =====================================================================
// weight prep: dst[Npad,K](bf16) = transpose(src[K,N](f32)), zero pad
// =====================================================================
__global__ __launch_bounds__(256) void cvt_transpose_kernel(
    const float* __restrict__ src, __bf16* __restrict__ dst,
    int K, int N, int Npad)
{
  const int idx = blockIdx.x * 256 + threadIdx.x;
  if (idx >= Npad * K) return;
  const int n = idx / K;
  const int k = idx - n * K;
  const float v = (n < N) ? src[(long)k * N + n] : 0.0f;
  dst[idx] = (__bf16)v;
}

// plain f32 -> bf16
__global__ __launch_bounds__(256) void cvt_kernel(
    const float* __restrict__ src, __bf16* __restrict__ dst, int n)
{
  const int idx = blockIdx.x * 256 + threadIdx.x;
  if (idx < n) dst[idx] = (__bf16)src[idx];
}

// =====================================================================
// causal depthwise conv (width 4) over xBC channels + SiLU
// =====================================================================
__global__ __launch_bounds__(256) void conv_silu_kernel(
    const float* __restrict__ zx, const float* __restrict__ cw,
    const float* __restrict__ cb, float* __restrict__ xbc)
{
  const int idx = blockIdx.x * blockDim.x + threadIdx.x;
  if (idx >= MROWS * CONV_DIM) return;
  const int c  = idx % CONV_DIM;
  const int bl = idx / CONV_DIM;
  const int b  = bl / SEQLEN;
  const int l  = bl % SEQLEN;
  float accv = cb[c];
#pragma unroll
  for (int k = 0; k < D_CONV; ++k) {
    const int lp = l + k - (D_CONV - 1);
    if (lp >= 0)
      accv += zx[(long)(b * SEQLEN + lp) * ZXP + D_INNER + c] *
              cw[c * D_CONV + k];
  }
  xbc[idx] = accv / (1.0f + __expf(-accv));   // silu
}

// =====================================================================
// dt = softplus(dt_raw + dt_bias); decay = exp(-exp(A_log)*dt)
// =====================================================================
__global__ __launch_bounds__(256) void dt_decay_kernel(
    const float* __restrict__ zx, const float* __restrict__ dtb,
    const float* __restrict__ alog, float* __restrict__ dt,
    float* __restrict__ dec)
{
  const int idx = blockIdx.x * blockDim.x + threadIdx.x;
  if (idx >= MROWS * NHEADS) return;
  const int h  = idx % NHEADS;
  const int bl = idx / NHEADS;
  const float x  = zx[(long)bl * ZXP + D_INNER + CONV_DIM + h] + dtb[h];
  const float sp = (x > 20.0f) ? x : log1pf(__expf(x));
  dt[idx]  = sp;
  dec[idx] = __expf(-__expf(alog[h]) * sp);
}

// =====================================================================
// sequential selective scan: one block per (batch, head).
// thread owns p = tid/4, n-chunk = (tid%4)*16; 64x64 state in regs.
// cross-lane reduce over 4 n-chunks via wave32 shuffles; no barriers.
// =====================================================================
__global__ __launch_bounds__(256) void scan_kernel(
    const float* __restrict__ xbc, const float* __restrict__ dt,
    const float* __restrict__ dec, const float* __restrict__ Dp,
    float* __restrict__ ys)
{
  const int bh = blockIdx.x;        // 0..B*NHEADS-1
  const int b  = bh >> 5;
  const int h  = bh & 31;
  const int tid = threadIdx.x;
  const int p  = tid >> 2;          // 0..63
  const int q  = tid & 3;           // n-chunk
  const int n0 = q * 16;

  float s[16];
#pragma unroll
  for (int j = 0; j < 16; ++j) s[j] = 0.0f;
  const float Dv = Dp[h];

  for (int t = 0; t < SEQLEN; ++t) {
    const long row = (long)(b * SEQLEN + t);
    const float d    = dec[row * NHEADS + h];
    const float dtv  = dt[row * NHEADS + h];
    const float xv   = xbc[row * CONV_DIM + h * HEADDIM + p];
    const float* Bp  = xbc + row * CONV_DIM + D_INNER + n0;
    const float* Cp  = xbc + row * CONV_DIM + D_INNER + D_STATE + n0;
    __builtin_prefetch(Bp + CONV_DIM, 0, 1);   // next timestep row
    const float coef = dtv * xv;
    float part = 0.0f;
#pragma unroll
    for (int j = 0; j < 16; ++j) {
      s[j] = s[j] * d + coef * Bp[j];
      part += s[j] * Cp[j];
    }
    part += __shfl_xor(part, 1);
    part += __shfl_xor(part, 2);
    if (q == 0)
      ys[row * D_INNER + h * HEADDIM + p] = part + Dv * xv;
  }
}

// =====================================================================
// y = ys * silu(z); y *= rsqrt(mean(y^2)+eps) * norm_w  -> bf16 out
// =====================================================================
__global__ __launch_bounds__(256) void gate_rmsnorm_kernel(
    const float* __restrict__ ys, const float* __restrict__ zx,
    const float* __restrict__ nw, __bf16* __restrict__ out)
{
  const int row = blockIdx.x;
  const int tid = threadIdx.x;
  float v[8];
  float ss = 0.0f;
#pragma unroll
  for (int i = 0; i < 8; ++i) {
    const int c  = tid + i * 256;
    const float z  = zx[(long)row * ZXP + c];
    const float sz = z / (1.0f + __expf(-z));
    const float y  = ys[(long)row * D_INNER + c] * sz;
    v[i] = y;
    ss += y * y;
  }
  ss += __shfl_xor(ss, 1);
  ss += __shfl_xor(ss, 2);
  ss += __shfl_xor(ss, 4);
  ss += __shfl_xor(ss, 8);
  ss += __shfl_xor(ss, 16);
  __shared__ float red[8];
  if ((tid & 31) == 0) red[tid >> 5] = ss;
  __syncthreads();
  float tot = 0.0f;
#pragma unroll
  for (int i = 0; i < 8; ++i) tot += red[i];
  const float scale = rsqrtf(tot * (1.0f / D_INNER) + 1e-5f);
#pragma unroll
  for (int i = 0; i < 8; ++i) {
    const int c = tid + i * 256;
    out[(long)row * D_INNER + c] = (__bf16)(v[i] * scale * nw[c]);
  }
}

// =====================================================================
// host-side pipeline
// =====================================================================
static inline void launch_gemm(const __bf16* A, const __bf16* Wt,
                               const float* b, float* Cf, __bf16* Cb,
                               int M, int Npad, int K, int act,
                               hipStream_t stream)
{
  dim3 grid(Npad / BN, M / BM);
  gemm_wmma<<<grid, 256, 0, stream>>>(A, Wt, b, Cf, Cb, M, Npad, K, act);
}

static inline void launch_cvtT(const float* src, __bf16* dst, int K, int N,
                               int Npad, hipStream_t stream)
{
  const int total = Npad * K;
  cvt_transpose_kernel<<<(total + 255) / 256, 256, 0, stream>>>(
      src, dst, K, N, Npad);
}

extern "C" void kernel_launch(void* const* d_in, const int* in_sizes, int n_in,
                              void* d_out, int out_size, void* d_ws, size_t ws_size,
                              hipStream_t stream)
{
  (void)in_sizes; (void)n_in; (void)out_size; (void)ws_size;

  // input order: x, mlp_in{W1,b1,W2,b2}, 4x blocks{in_proj,conv_w,conv_b,
  // dt_bias,A_log,D,norm_w,out_proj}, mlp_out{W1,b1,W2,b2}
  const float* x    = (const float*)d_in[0];
  const float* miW1 = (const float*)d_in[1];
  const float* mib1 = (const float*)d_in[2];
  const float* miW2 = (const float*)d_in[3];
  const float* mib2 = (const float*)d_in[4];
  const float* moW1 = (const float*)d_in[37];
  const float* mob1 = (const float*)d_in[38];
  const float* moW2 = (const float*)d_in[39];
  const float* mob2 = (const float*)d_in[40];

  // workspace carve-up (byte offsets, all 16B aligned)
  char* wsb = (char*)d_ws;
  __bf16* wsc = (__bf16*)wsb; wsb += (size_t)ZXP * 1024 * 2;     // weight scratch (max)
  __bf16* xbf = (__bf16*)wsb; wsb += (size_t)MROWS * D_MODEL * 2;
  __bf16* t0b = (__bf16*)wsb; wsb += (size_t)MROWS * D_MODEL * 2;
  __bf16* h0b = (__bf16*)wsb; wsb += (size_t)MROWS * D_MODEL * 2;
  __bf16* h1b = (__bf16*)wsb; wsb += (size_t)MROWS * D_MODEL * 2;
  __bf16* ynb = (__bf16*)wsb; wsb += (size_t)MROWS * D_INNER * 2;
  float*  zx  = (float*)wsb;  wsb += (size_t)MROWS * ZXP * 4;
  float*  xbc = (float*)wsb;  wsb += (size_t)MROWS * CONV_DIM * 4;
  float*  dtb = (float*)wsb;  wsb += (size_t)MROWS * NHEADS * 4;
  float*  dcb = (float*)wsb;  wsb += (size_t)MROWS * NHEADS * 4;
  float*  ysb = (float*)wsb;  wsb += (size_t)MROWS * D_INNER * 4;

  // ---- input MLP ----
  cvt_kernel<<<(MROWS * STATE_SZ + 255) / 256, 256, 0, stream>>>(
      x, xbf, MROWS * STATE_SZ);
  launch_cvtT(miW1, wsc, STATE_SZ, D_MODEL, D_MODEL, stream);
  launch_gemm(xbf, wsc, mib1, nullptr, t0b, MROWS, D_MODEL, STATE_SZ, 1, stream);
  launch_cvtT(miW2, wsc, D_MODEL, D_MODEL, D_MODEL, stream);
  launch_gemm(t0b, wsc, mib2, nullptr, h0b, MROWS, D_MODEL, D_MODEL, 0, stream);

  // ---- 4 Mamba2 blocks ----
  __bf16* hcur  = h0b;
  __bf16* hnext = h1b;
  for (int blk = 0; blk < 4; ++blk) {
    const int base = 5 + blk * 8;
    const float* inp    = (const float*)d_in[base + 0];
    const float* cw     = (const float*)d_in[base + 1];
    const float* cb     = (const float*)d_in[base + 2];
    const float* dtbias = (const float*)d_in[base + 3];
    const float* alog   = (const float*)d_in[base + 4];
    const float* Dp     = (const float*)d_in[base + 5];
    const float* nw     = (const float*)d_in[base + 6];
    const float* outp   = (const float*)d_in[base + 7];

    launch_cvtT(inp, wsc, D_MODEL, D_IN_PROJ, ZXP, stream);
    launch_gemm(hcur, wsc, nullptr, zx, nullptr, MROWS, ZXP, D_MODEL, 0, stream);
    conv_silu_kernel<<<(MROWS * CONV_DIM + 255) / 256, 256, 0, stream>>>(
        zx, cw, cb, xbc);
    dt_decay_kernel<<<(MROWS * NHEADS + 255) / 256, 256, 0, stream>>>(
        zx, dtbias, alog, dtb, dcb);
    scan_kernel<<<BATCH * NHEADS, 256, 0, stream>>>(xbc, dtb, dcb, Dp, ysb);
    gate_rmsnorm_kernel<<<MROWS, 256, 0, stream>>>(ysb, zx, nw, ynb);
    launch_cvtT(outp, wsc, D_INNER, D_MODEL, D_MODEL, stream);
    launch_gemm(ynb, wsc, nullptr, nullptr, hnext, MROWS, D_MODEL, D_INNER, 0, stream);

    __bf16* tmp = hcur; hcur = hnext; hnext = tmp;
  }

  // ---- output MLP (+ fused sigmoid into d_out) ----
  launch_cvtT(moW1, wsc, D_MODEL, 256, 256, stream);
  launch_gemm(hcur, wsc, mob1, nullptr, t0b, MROWS, 256, D_MODEL, 1, stream);
  launch_cvtT(moW2, wsc, 256, 256, 256, stream);
  launch_gemm(t0b, wsc, mob2, (float*)d_out, nullptr, MROWS, 256, 256, 2, stream);
}
